// CONV_LSTM_87033217286791
// MI455X (gfx1250) — matrix operand order repaired and checked by reading, not compiled, not run on hardware
//
#include <hip/hip_runtime.h>

// ---------------------------------------------------------------------------
// Peephole Conv-LSTM (T=8, B=8, C=64, H=W=96, L=2, K=3) for gfx1250.
// Implicit-GEMM on bf16 WMMA. Activations in bf16 channel-last padded buffers
// [B][98][98][64] (zero borders). Both input tiles AND per-tap weight tiles
// are streamed into LDS with global_load_async_to_lds_b128 (double-buffered),
// so all WMMA operands come from contiguous LDS b128 reads.
// ---------------------------------------------------------------------------

typedef __bf16  v16bf __attribute__((ext_vector_type(16)));
typedef float   v8f   __attribute__((ext_vector_type(8)));
typedef unsigned short u16v8 __attribute__((ext_vector_type(8)));

static constexpr int kT = 8, kB = 8, kC = 64, kH = 96, kW = 96, kL = 2;
static constexpr int kHP = 98, kWP = 98;          // padded spatial
static constexpr int kK = 576;                    // Cin*9, k = tap*64 + cin
static constexpr int TW = 16, TH = 8;             // 128 px per workgroup
static constexpr int HALO_W = TW + 2, HALO_H = TH + 2;   // 18 x 10 cells
static constexpr int NCELL = HALO_W * HALO_H;            // 180
static constexpr int CELLS = 72;                  // ushorts per cell (64+8 pad)
static constexpr int TILE_USH = NCELL * CELLS;    // 12,960 ushorts
static constexpr int TILE_B   = TILE_USH * 2;     // 25,920 B
// weight-tap staging: frag f=(g*4+mt)*2+c0h, each frag = 32 lanes x 32B
static constexpr int WTAP_USH_G = 4 * 8 * 512;    // 4 gates  -> 16,384 ushorts
static constexpr int WTAP_B_G   = WTAP_USH_G * 2; // 32,768 B
static constexpr int WTAP_USH_O = 1 * 8 * 512;    // 1 gate   -> 4,096 ushorts
static constexpr int WTAP_B_O   = WTAP_USH_O * 2; // 8,192 B
static constexpr size_t GATES_SMEM = (size_t)2 * TILE_B + 2 * WTAP_B_G; // 117,376
static constexpr size_t OUT_SMEM   = (size_t)TILE_B + 2 * WTAP_B_O;     //  42,304
// per padded image / buffer sizes (bf16 elements)
static constexpr size_t PIMG = (size_t)kHP * kWP * kC;        // 614,656
static constexpr size_t PBUF = (size_t)kB * PIMG;             // 4,917,248

__device__ __forceinline__ unsigned short f2bf(float f) {
  unsigned u = __builtin_bit_cast(unsigned, f);
  unsigned r = u + 0x7FFFu + ((u >> 16) & 1u);
  return (unsigned short)(r >> 16);
}
__device__ __forceinline__ float bf2f(unsigned short b) {
  return __builtin_bit_cast(float, (unsigned)b << 16);
}
__device__ __forceinline__ float sigmoidf_(float x) {
  return 1.0f / (1.0f + __expf(-x));
}

// Async 16B global -> LDS copy (ASYNCcnt-tracked; per-lane LDS address).
__device__ __forceinline__ void async_cp16(unsigned lds_off, const void* gaddr) {
  asm volatile("global_load_async_to_lds_b128 %0, %1, off"
               :: "v"(lds_off), "v"(gaddr) : "memory");
}
__device__ __forceinline__ void async_wait0() {
  asm volatile("s_wait_asynccnt 0x0" ::: "memory");
}

// A fragment from LDS weight-tap buffer: frag f = (g*4+mt)*2+c0h,
// lane data contiguous 32B at f*512 + lane*16 (ushort units).
__device__ __forceinline__ v16bf load_afrag_lds(const unsigned short* __restrict__ wt,
                                                int f, int lane) {
  const uint4* p = (const uint4*)(wt + f * 512 + lane * 16);
  struct AB { uint4 a, b; } t;
  t.a = p[0];
  t.b = p[1];
  return __builtin_bit_cast(v16bf, t);
}

// B fragment: 32x16 bf16. Channel-last LDS tile -> lane (n,kh) reads 16
// consecutive cin values: one contiguous 32B LDS read (2 x b128).
__device__ __forceinline__ v16bf load_bfrag(const unsigned short* __restrict__ lds,
                                            int cell, int c0kh) {
  const uint4* p = (const uint4*)(lds + cell * CELLS + c0kh);
  struct AB { uint4 a, b; } t;
  t.a = p[0];
  t.b = p[1];
  return __builtin_bit_cast(v16bf, t);
}

__device__ __forceinline__ int jsel(int s, int g) {
  int jx = (g == 0) ? 0 : (g == 1) ? 3 : (g == 2) ? 6 : 8;  // wxi wxf wxc wxo
  int jh = (g == 0) ? 1 : (g == 1) ? 4 : (g == 2) ? 7 : 9;  // whi whf whc who
  int jc = (g == 0) ? 2 : 5;                                // wci wcf
  return (s == 0) ? jx : (s == 1) ? jh : jc;
}

// Issue async staging of one padded-NHWC source tile to LDS byte offset ldsoff.
__device__ __forceinline__ void stage_tile(const unsigned short* __restrict__ sp,
                                           unsigned ldsoff,
                                           int b, int ty, int tx, int tid) {
  for (int i = tid; i < NCELL * 8; i += 256) {     // 1440 x 16B chunks
    int cell = i >> 3, q = i & 7;
    int r = cell / HALO_W, cc = cell % HALO_W;
    const unsigned short* g =
        sp + (((size_t)b * kHP + ty + r) * kWP + (tx + cc)) * kC + q * 8;
    async_cp16(ldsoff + (unsigned)(cell * CELLS * 2 + q * 16), g);
  }
}

// Issue async staging of one tap's A-fragments (ng gates) to LDS.
// chunk i -> piece(i&1), lane((i>>1)&31), frag f(i>>6); f = (g*4+mt)*2+c0h.
__device__ __forceinline__ void stage_wtap(const unsigned short* __restrict__ w0,
                                           const unsigned short* __restrict__ w1,
                                           const unsigned short* __restrict__ w2,
                                           const unsigned short* __restrict__ w3,
                                           int ng, int tap, unsigned ldsoff, int tid) {
  const int nchunk = ng * 512;
  for (int i = tid; i < nchunk; i += 256) {
    int piece = i & 1;
    int lane2 = (i >> 1) & 31;
    int f = i >> 6;
    int c0h = f & 1, mt = (f >> 1) & 3, g = f >> 3;
    const unsigned short* wgp = (g == 0) ? w0 : (g == 1) ? w1 : (g == 2) ? w2 : w3;
    int m = mt * 16 + (lane2 & 15);
    int kh = lane2 >> 4;
    const unsigned short* src =
        wgp + (size_t)m * kK + tap * 64 + c0h * 32 + 16 * kh + piece * 8;
    async_cp16(ldsoff + (unsigned)(f * 1024 + lane2 * 32 + piece * 16), src);
  }
}

// ---------------------------------------------------------------------------
// Pack f32 [L,C,C,3,3] weights into bf16 [j][l][cout][k=tap*64+cin]
// ---------------------------------------------------------------------------
__global__ void __launch_bounds__(256)
clstm_pack_w(const float* __restrict__ w0, const float* __restrict__ w1,
             const float* __restrict__ w2, const float* __restrict__ w3,
             const float* __restrict__ w4, const float* __restrict__ w5,
             const float* __restrict__ w6, const float* __restrict__ w7,
             const float* __restrict__ w8, const float* __restrict__ w9,
             const float* __restrict__ w10, unsigned short* __restrict__ wbf) {
  int idx = blockIdx.x * 256 + threadIdx.x;
  const int per = kL * kC * kK;
  if (idx >= 11 * per) return;
  int j = idx / per, r = idx % per;
  int l = r / (kC * kK); r %= kC * kK;
  int cout = r / kK;
  int k = r % kK;
  int tap = k / 64, cin = k % 64;
  int ky = tap / 3, kx = tap % 3;
  const float* w = (j == 0) ? w0 : (j == 1) ? w1 : (j == 2) ? w2 : (j == 3) ? w3
                 : (j == 4) ? w4 : (j == 5) ? w5 : (j == 6) ? w6 : (j == 7) ? w7
                 : (j == 8) ? w8 : (j == 9) ? w9 : w10;
  float v = w[((((size_t)l * kC + cout) * kC + cin) * 3 + ky) * 3 + kx];
  wbf[idx] = f2bf(v);
}

__global__ void __launch_bounds__(256)
clstm_zero(uint4* __restrict__ p, long long n16) {
  long long i = (long long)blockIdx.x * 256 + threadIdx.x;
  long long stride = (long long)gridDim.x * 256;
  uint4 z; z.x = z.y = z.z = z.w = 0u;
  for (; i < n16; i += stride) p[i] = z;
}

// Convert x: f32 NCHW -> bf16 padded NHWC [T][B][98][98][64]
__global__ void __launch_bounds__(256)
clstm_cvt_x(const float* __restrict__ x, unsigned short* __restrict__ xbf) {
  long long idx = (long long)blockIdx.x * 256 + threadIdx.x;
  const long long total = (long long)kT * PBUF;
  long long stride = (long long)gridDim.x * 256;
  for (; idx < total; idx += stride) {
    int ch = (int)(idx & 63);
    long long rest = idx >> 6;
    int xp = (int)(rest % kWP); rest /= kWP;
    int yp = (int)(rest % kHP); rest /= kHP;
    int b = (int)(rest % kB);
    long long t = rest / kB;
    float v = 0.0f;
    if (xp >= 1 && xp <= kW && yp >= 1 && yp <= kH)
      v = x[(((t * kB + b) * kC + ch) * (size_t)kH + (yp - 1)) * kW + (xp - 1)];
    xbf[idx] = f2bf(v);
  }
}

// ---------------------------------------------------------------------------
// Kernel 1: gates. Sources {x,h,c}; double-buffered async staging of both
// input tiles and per-tap weight fragments. Writes c_new (bf16 NHWC padded,
// + f32 NCHW if wf32) and o_pre (f32 NCHW).
// ---------------------------------------------------------------------------
__global__ void __launch_bounds__(256)
clstm_gates(const unsigned short* __restrict__ xt,
            const unsigned short* __restrict__ hprev,
            const unsigned short* __restrict__ cprev,
            const unsigned short* __restrict__ wbf,
            const float* __restrict__ b_i, const float* __restrict__ b_f,
            const float* __restrict__ b_c, const float* __restrict__ b_o,
            unsigned short* __restrict__ cnbf, float* __restrict__ cnf32,
            float* __restrict__ opre, int layer, int has_state, int wf32) {
  extern __shared__ unsigned short smem[];   // [in0][in1][wtap0][wtap1]

  const int tid = threadIdx.x;
  const int lane = tid & 31, wv = tid >> 5;
  const int n = lane & 15, half = lane >> 4;
  const int tx = blockIdx.x * TW, ty = blockIdx.y * TH, b = blockIdx.z;
  const unsigned lds0 = (unsigned)(uintptr_t)(void*)smem;

  v8f acc[4][4];
#pragma unroll
  for (int g = 0; g < 4; ++g)
#pragma unroll
    for (int mt = 0; mt < 4; ++mt) acc[g][mt] = (v8f)(0.0f);

  const int nsrc = has_state ? 3 : 1;

  stage_tile(xt, lds0, b, ty, tx, tid);

  for (int s = 0; s < nsrc; ++s) {
    const unsigned short* in = smem + (s & 1) * TILE_USH;
    const int ngates = (s == 2) ? 2 : 4;
    const unsigned short* wg0 = wbf + (size_t)(jsel(s, 0) * kL + layer) * kC * kK;
    const unsigned short* wg1 = wbf + (size_t)(jsel(s, 1) * kL + layer) * kC * kK;
    const unsigned short* wg2 = wbf + (size_t)(jsel(s, 2) * kL + layer) * kC * kK;
    const unsigned short* wg3 = wbf + (size_t)(jsel(s, 3) * kL + layer) * kC * kK;

    stage_wtap(wg0, wg1, wg2, wg3, ngates, 0, lds0 + 2 * TILE_B, tid);
    async_wait0();
    __syncthreads();

    for (int tap = 0; tap < 9; ++tap) {
      if (tap == 0 && s + 1 < nsrc)   // overlap next source's input tile
        stage_tile(s == 0 ? hprev : cprev,
                   lds0 + (unsigned)(((s + 1) & 1) * TILE_B), b, ty, tx, tid);
      if (tap + 1 < 9)                // overlap next tap's weight fragments
        stage_wtap(wg0, wg1, wg2, wg3, ngates, tap + 1,
                   lds0 + 2 * TILE_B + (unsigned)(((tap + 1) & 1) * WTAP_B_G), tid);

      const unsigned short* wt = smem + 2 * TILE_USH + (tap & 1) * WTAP_USH_G;
      const int ky = tap / 3, kx = tap % 3;
      const int cell = (wv + ky) * HALO_W + (n + kx);
#pragma unroll
      for (int c0h = 0; c0h < 2; ++c0h) {
        v16bf bfrag = load_bfrag(in, cell, c0h * 32 + 16 * half);
#pragma unroll
        for (int g = 0; g < 4; ++g) {
          if (g >= ngates) break;
#pragma unroll
          for (int mt = 0; mt < 4; ++mt) {
            v16bf afrag = load_afrag_lds(wt, (g * 4 + mt) * 2 + c0h, lane);
            acc[g][mt] = __builtin_amdgcn_wmma_f32_16x16x32_bf16(
                false, afrag, false, bfrag, (short)0, acc[g][mt], false, false);
          }
        }
      }
      async_wait0();
      __syncthreads();
    }
  }

  // Epilogue. c-source tile (s=2) sits in input buffer 0.
  const int y = ty + wv, xx = tx + n;
  const size_t pixf = (((size_t)b * kC) * kH + y) * kW + xx;             // NCHW
  const size_t pixp = (((size_t)b * kHP + y + 1) * kWP + (xx + 1)) * kC; // NHWC
  const int ccell = (wv + 1) * HALO_W + (n + 1);
#pragma unroll
  for (int mt = 0; mt < 4; ++mt) {
    u16v8 pk;
#pragma unroll
    for (int v = 0; v < 8; ++v) {
      int ch = mt * 16 + v + half * 8;
      float ai = acc[0][mt][v] + b_i[layer * kC + ch];
      float af = acc[1][mt][v] + b_f[layer * kC + ch];
      float ac = acc[2][mt][v] + b_c[layer * kC + ch];
      float ao = acc[3][mt][v] + b_o[layer * kC + ch];
      float cp = 0.0f;
      if (has_state) cp = bf2f(smem[ccell * CELLS + ch]);
      float cn = tanhf(ac) * sigmoidf_(ai) + cp * sigmoidf_(af);
      pk[v] = f2bf(cn);
      opre[pixf + (size_t)ch * kH * kW] = ao;
      if (wf32) cnf32[pixf + (size_t)ch * kH * kW] = cn;
    }
    *(uint4*)(cnbf + pixp + mt * 16 + half * 8) = __builtin_bit_cast(uint4, pk);
  }
}

// ---------------------------------------------------------------------------
// Kernel 2: o = sigmoid(o_pre + conv(c_new, Whc) + b_c); h = o * tanh(c_new)
// ---------------------------------------------------------------------------
__global__ void __launch_bounds__(256)
clstm_out(const unsigned short* __restrict__ cnbf, const float* __restrict__ opre,
          const unsigned short* __restrict__ wbf, const float* __restrict__ b_c,
          unsigned short* __restrict__ hbf, float* __restrict__ hf32,
          int layer, int wf32) {
  extern __shared__ unsigned short smem[];   // [in][wtap0][wtap1]

  const int tid = threadIdx.x;
  const int lane = tid & 31, wv = tid >> 5;
  const int n = lane & 15, half = lane >> 4;
  const int tx = blockIdx.x * TW, ty = blockIdx.y * TH, b = blockIdx.z;
  const unsigned lds0 = (unsigned)(uintptr_t)(void*)smem;

  const unsigned short* whc = wbf + (size_t)(7 * kL + layer) * kC * kK;  // j=7

  stage_tile(cnbf, lds0, b, ty, tx, tid);
  stage_wtap(whc, whc, whc, whc, 1, 0, lds0 + TILE_B, tid);
  async_wait0();
  __syncthreads();

  v8f acc[4];
#pragma unroll
  for (int mt = 0; mt < 4; ++mt) acc[mt] = (v8f)(0.0f);

  for (int tap = 0; tap < 9; ++tap) {
    if (tap + 1 < 9)
      stage_wtap(whc, whc, whc, whc, 1, tap + 1,
                 lds0 + TILE_B + (unsigned)(((tap + 1) & 1) * WTAP_B_O), tid);
    const unsigned short* wt = smem + TILE_USH + (tap & 1) * WTAP_USH_O;
    const int ky = tap / 3, kx = tap % 3;
    const int cell = (wv + ky) * HALO_W + (n + kx);
#pragma unroll
    for (int c0h = 0; c0h < 2; ++c0h) {
      v16bf bfrag = load_bfrag(smem, cell, c0h * 32 + 16 * half);
#pragma unroll
      for (int mt = 0; mt < 4; ++mt) {
        v16bf afrag = load_afrag_lds(wt, mt * 2 + c0h, lane);
        acc[mt] = __builtin_amdgcn_wmma_f32_16x16x32_bf16(
            false, afrag, false, bfrag, (short)0, acc[mt], false, false);
      }
    }
    async_wait0();
    __syncthreads();
  }

  const int y = ty + wv, xx = tx + n;
  const size_t pixf = (((size_t)b * kC) * kH + y) * kW + xx;
  const size_t pixp = (((size_t)b * kHP + y + 1) * kWP + (xx + 1)) * kC;
  const int ccell = (wv + 1) * HALO_W + (n + 1);
#pragma unroll
  for (int mt = 0; mt < 4; ++mt) {
    u16v8 pk;
#pragma unroll
    for (int v = 0; v < 8; ++v) {
      int ch = mt * 16 + v + half * 8;
      float cn = bf2f(smem[ccell * CELLS + ch]);
      float o = sigmoidf_(opre[pixf + (size_t)ch * kH * kW] + acc[mt][v]
                          + b_c[layer * kC + ch]);
      float h = o * tanhf(cn);
      pk[v] = f2bf(h);
      if (wf32) hf32[pixf + (size_t)ch * kH * kW] = h;
    }
    *(uint4*)(hbf + pixp + mt * 16 + half * 8) = __builtin_bit_cast(uint4, pk);
  }
}

// ---------------------------------------------------------------------------
extern "C" void kernel_launch(void* const* d_in, const int* in_sizes, int n_in,
                              void* d_out, int out_size, void* d_ws, size_t ws_size,
                              hipStream_t stream) {
  (void)in_sizes; (void)n_in; (void)out_size; (void)ws_size;

  const float* x = (const float*)d_in[0];
  const float* w[11];
  for (int i = 0; i < 11; ++i) w[i] = (const float*)d_in[1 + i];
  const float* b_i = (const float*)d_in[12];
  const float* b_f = (const float*)d_in[13];
  const float* b_c = (const float*)d_in[14];
  const float* b_o = (const float*)d_in[15];

  const size_t S = (size_t)kB * kC * kH * kW;     // per-step f32 state elems
  float* hs = (float*)d_out;
  float* cs = hs + (size_t)kT * S;

  char* ws = (char*)d_ws;
  unsigned short* wbf = (unsigned short*)ws;                  // 1.62 MB
  size_t off = (size_t)2 << 20;
  float* opre = (float*)(ws + off); off += S * sizeof(float); // 18.9 MB
  float* fscr = (float*)(ws + off); off += S * sizeof(float); // f32 dump (l=0)
  unsigned short* xbf  = (unsigned short*)(ws + off); off += kT * PBUF * 2;
  unsigned short* hbf0 = (unsigned short*)(ws + off); off += kT * PBUF * 2;
  unsigned short* hbf1 = (unsigned short*)(ws + off); off += 2 * PBUF * 2;
  unsigned short* cbf  = (unsigned short*)(ws + off); off += 2 * PBUF * 2;

  {
    const int tot = 11 * kL * kC * kK;
    clstm_pack_w<<<(tot + 255) / 256, 256, 0, stream>>>(
        w[0], w[1], w[2], w[3], w[4], w[5], w[6], w[7], w[8], w[9], w[10], wbf);
    long long n16 = (long long)(12 * PBUF * 2) / 16;   // hbf0+hbf1+cbf
    clstm_zero<<<2048, 256, 0, stream>>>((uint4*)hbf0, n16);
    clstm_cvt_x<<<4096, 256, 0, stream>>>(x, xbf);
  }

  dim3 grid(kW / TW, kH / TH, kB);   // 6 x 12 x 8
  dim3 block(256);

  for (int l = 0; l < kL; ++l) {
    for (int t = 0; t < kT; ++t) {
      const unsigned short* xtp = (l == 0) ? xbf + (size_t)t * PBUF
                                           : hbf0 + (size_t)t * PBUF;
      const unsigned short* hp =
          (t == 0) ? xtp
                   : (l == 0 ? hbf0 + (size_t)(t - 1) * PBUF
                             : hbf1 + (size_t)((t - 1) & 1) * PBUF);
      const unsigned short* cpv = (t == 0) ? xtp
                                           : cbf + (size_t)((t - 1) & 1) * PBUF;
      unsigned short* cnb = cbf + (size_t)(t & 1) * PBUF;
      float* cnf = (l == 0) ? fscr : cs + (size_t)t * S;
      clstm_gates<<<grid, block, GATES_SMEM, stream>>>(xtp, hp, cpv, wbf,
                                                       b_i, b_f, b_c, b_o,
                                                       cnb, cnf, opre, l,
                                                       t > 0 ? 1 : 0,
                                                       l == 1 ? 1 : 0);
      unsigned short* hbp = (l == 0) ? hbf0 + (size_t)t * PBUF
                                     : hbf1 + (size_t)(t & 1) * PBUF;
      float* hfp = (l == 0) ? fscr : hs + (size_t)t * S;
      clstm_out<<<grid, block, OUT_SMEM, stream>>>(cnb, opre, wbf, b_c, hbp, hfp,
                                                   l, l == 1 ? 1 : 0);
    }
  }
}
